// AttentionLayer_37804302139596
// MI455X (gfx1250) — matrix operand order repaired and checked by reading, not compiled, not run on hardware
//
#include <hip/hip_runtime.h>
#include <hip/hip_bf16.h>

#define HID   1024
#define NH    16
#define HD    64
#define SEQ   2048
#define BATCH 4
#define NTOK  (BATCH * SEQ) /* 8192 */

typedef __bf16 bf16_t;
typedef __attribute__((ext_vector_type(16))) __bf16 v16bf;
typedef __attribute__((ext_vector_type(8)))  __bf16 v8bf;
typedef __attribute__((ext_vector_type(4)))  __bf16 v4bf;
typedef __attribute__((ext_vector_type(8)))  float  v8f;
typedef __attribute__((ext_vector_type(4)))  float  v4f;

static __device__ __forceinline__ v8f wmma_bf16(v16bf a, v16bf b, v8f c) {
  // D = A(16x32 bf16) * B(32x16 bf16) + C(16x16 f32)
  return __builtin_amdgcn_wmma_f32_16x16x32_bf16(false, a, false, b, (short)0, c, false, false);
}

// A-fragment (16x32 bf16): lane holds K = half*16+{0..7} in elems 0..7 and
// K = half*16+8+{0..7} in elems 8..15 (contiguous order, matches B loads).
static __device__ __forceinline__ v16bf make_a(const bf16_t* __restrict__ row, int k0, int half) {
  const v8bf lo = *(const v8bf*)(row + k0 + half * 16);
  const v8bf hi = *(const v8bf*)(row + k0 + half * 16 + 8);
  v16bf r;
#pragma unroll
  for (int i = 0; i < 8; ++i) { r[i] = lo[i]; r[8 + i] = hi[i]; }
  return r;
}

static __device__ __forceinline__ v8f vzero8() {
  v8f z = {0.f, 0.f, 0.f, 0.f, 0.f, 0.f, 0.f, 0.f};
  return z;
}

// Fully-unrolled GEMM core: 32(M) x 64(N) wave tile over K = HID.
// Straight-line SSA (no loop backedge) so accumulators never need phi
// copies; the scheduler hoists next-step loads as deep as pressure allows.
#define GEMM_CORE()                                                             \
  v8f c00 = vzero8(), c01 = vzero8(), c02 = vzero8(), c03 = vzero8();           \
  v8f c10 = vzero8(), c11 = vzero8(), c12 = vzero8(), c13 = vzero8();           \
  _Pragma("unroll")                                                             \
  for (int k0 = 0; k0 < HID; k0 += 32) {                                        \
    const v16bf a0 = make_a(arow0, k0, half);                                   \
    const v16bf a1 = make_a(arow1, k0, half);                                   \
    const v16bf b0 = *(const v16bf*)(brow0 + (size_t)0 * 16 * HID + k0 + half * 16); \
    const v16bf b1 = *(const v16bf*)(brow0 + (size_t)1 * 16 * HID + k0 + half * 16); \
    const v16bf b2 = *(const v16bf*)(brow0 + (size_t)2 * 16 * HID + k0 + half * 16); \
    const v16bf b3 = *(const v16bf*)(brow0 + (size_t)3 * 16 * HID + k0 + half * 16); \
    c00 = wmma_bf16(a0, b0, c00);                                               \
    c10 = wmma_bf16(a1, b0, c10);                                               \
    c01 = wmma_bf16(a0, b1, c01);                                               \
    c11 = wmma_bf16(a1, b1, c11);                                               \
    c02 = wmma_bf16(a0, b2, c02);                                               \
    c12 = wmma_bf16(a1, b2, c12);                                               \
    c03 = wmma_bf16(a0, b3, c03);                                               \
    c13 = wmma_bf16(a1, b3, c13);                                               \
  }

// ---------------------------------------------------------------- converts
__global__ __launch_bounds__(256) void f32_to_bf16_kernel(const v4f* __restrict__ src,
                                                          v4bf* __restrict__ dst, int n4) {
  const int i = blockIdx.x * 256 + threadIdx.x;
  if (i < n4) {
    const v4f x = src[i];
    v4bf y;
#pragma unroll
    for (int j = 0; j < 4; ++j) y[j] = (bf16_t)x[j];
    dst[i] = y;
  }
}

// ---------------------------------------------------------------- QKV GEMM
// q = X @ W^T + b (NT GEMM). grid(16, 32, 3), block 256 = 8 waves; wave
// computes 32x64. mat 2 (V) is stored transposed: Vt[b][h][d][s].
__global__ __launch_bounds__(256)
void qkv_gemm_kernel(const bf16_t* __restrict__ Xb,
                     const bf16_t* __restrict__ Wqb, const bf16_t* __restrict__ Wkb,
                     const bf16_t* __restrict__ Wvb,
                     const float* __restrict__ bq, const float* __restrict__ bk,
                     const float* __restrict__ bv,
                     bf16_t* __restrict__ Qb, bf16_t* __restrict__ Kb,
                     bf16_t* __restrict__ Vtb) {
  const int lane = threadIdx.x & 31;
  const int wave = threadIdx.x >> 5;
  const int half = lane >> 4;
  const int lm   = lane & 15;
  const int n0   = blockIdx.x * 64;
  const int m0   = blockIdx.y * 256 + wave * 32;
  const int mat  = blockIdx.z;

  const bf16_t* W    = (mat == 0) ? Wqb : (mat == 1) ? Wkb : Wvb;
  const float*  bias = (mat == 0) ? bq  : (mat == 1) ? bk  : bv;
  bf16_t*       qk   = (mat == 0) ? Qb  : Kb;

  const bf16_t* arow0 = Xb + (size_t)(m0 + lm) * HID;
  const bf16_t* arow1 = arow0 + (size_t)16 * HID;
  const bf16_t* brow0 = W + (size_t)(n0 + lm) * HID;

  GEMM_CORE()

  auto store_one = [&](const v8f& c, int t, int ms) {
    const int n = n0 + t * 16 + lm;
    const float bia = bias[n];
#pragma unroll
    for (int r = 0; r < 8; ++r) {
      const int m = m0 + ms * 16 + half * 8 + r;
      const float v = c[r] + bia;
      if (mat == 2) {
        const int bb = m >> 11, s = m & 2047, hh = n >> 6, d = n & 63;
        Vtb[(((size_t)(bb * NH + hh) * HD + d) << 11) + s] = (bf16_t)v;
      } else {
        qk[(size_t)m * HID + n] = (bf16_t)v;
      }
    }
  };
  store_one(c00, 0, 0); store_one(c10, 0, 1);
  store_one(c01, 1, 0); store_one(c11, 1, 1);
  store_one(c02, 2, 0); store_one(c12, 2, 1);
  store_one(c03, 3, 0); store_one(c13, 3, 1);
}

// ---------------------------------------------------------------- attention
// One block = (16 queries) x (one head) x (one batch). 8 waves; wave w owns
// keys [w*256, w*256+256). Full 16x2048 f32 score strip lives in LDS
// (stride 2052 to stagger banks), softmax in LDS, probs streamed out NT,
// then P@V with cross-wave ds_add_f32 reduction. 136.5 KB LDS (<320 KB/WG).
__global__ __launch_bounds__(256)
void attn_kernel(const bf16_t* __restrict__ Qb, const bf16_t* __restrict__ Kb,
                 const bf16_t* __restrict__ Vtb, const float* __restrict__ amask,
                 float* __restrict__ probs, bf16_t* __restrict__ Ctxb) {
  __shared__ float s_sc[16 * 2052];
  __shared__ float s_red[16 * 16];
  __shared__ float s_inv[16];
  __shared__ float s_ctx[16 * 64];

  const int tid  = threadIdx.x;
  const int lane = tid & 31;
  const int wave = tid >> 5;
  const int half = lane >> 4;
  const int lm   = lane & 15;
  const int q0   = blockIdx.x * 16;
  const int h    = blockIdx.y;
  const int b    = blockIdx.z;

  for (int i = tid; i < 16 * 64; i += 256) s_ctx[i] = 0.0f;

  // ---- phase 1: scores = Q K^T / 8 + addmask  (NT GEMM over d=64)
  const bf16_t* qrow = Qb + (size_t)(b * SEQ + q0 + lm) * HID + h * HD;
  const v16bf aq0 = make_a(qrow, 0, half);
  const v16bf aq1 = make_a(qrow, 32, half);

  const bf16_t* kbase = Kb + (size_t)(b * SEQ + wave * 256 + lm) * HID + h * HD;

#pragma unroll
  for (int kt = 0; kt < 16; ++kt) {
    const bf16_t* kr = kbase + (size_t)kt * 16 * HID;
    const v16bf kb0 = *(const v16bf*)(kr + half * 16);
    const v16bf kb1 = *(const v16bf*)(kr + 32 + half * 16);
    v8f c = vzero8();
    c = wmma_bf16(aq0, kb0, c);
    c = wmma_bf16(aq1, kb1, c);
    const int j0 = wave * 256 + kt * 16;
    const float am = (1.0f - amask[b * SEQ + j0 + lm]) * -10000.0f;
#pragma unroll
    for (int r = 0; r < 8; ++r)
      s_sc[(half * 8 + r) * 2052 + j0 + lm] = c[r] * 0.125f + am;
  }
  __syncthreads();

  // ---- phase 2: row softmax over 2048 keys, 16 threads per row
  const int r    = tid & 15;
  const int cseg = tid >> 4;      // 0..15, owns 128 contiguous cols
  float pmax = -3.0e38f;
  for (int i = 0; i < 128; ++i)
    pmax = fmaxf(pmax, s_sc[r * 2052 + cseg * 128 + i]);
  s_red[r * 16 + cseg] = pmax;
  __syncthreads();
  float rmax = -3.0e38f;
#pragma unroll
  for (int i = 0; i < 16; ++i) rmax = fmaxf(rmax, s_red[r * 16 + i]);
  __syncthreads();
  float psum = 0.0f;
  for (int i = 0; i < 128; ++i) {
    const int idx = r * 2052 + cseg * 128 + i;
    const float e = __expf(s_sc[idx] - rmax);
    s_sc[idx] = e;
    psum += e;
  }
  s_red[r * 16 + cseg] = psum;
  __syncthreads();
  if (cseg == 0) {
    float s = 0.0f;
#pragma unroll
    for (int i = 0; i < 16; ++i) s += s_red[r * 16 + i];
    s_inv[r] = 1.0f / s;
  }
  __syncthreads();

  // normalize in LDS + stream probs to HBM (nontemporal: written once, never reread)
  float* gp = probs + ((size_t)(b * NH + h) * SEQ + q0) * SEQ;
  for (int idx = tid; idx < 16 * SEQ; idx += 256) {
    const int row = idx >> 11, col = idx & 2047;
    const float p = s_sc[row * 2052 + col] * s_inv[row];
    s_sc[row * 2052 + col] = p;
    __builtin_nontemporal_store(p, gp + idx);
  }
  __syncthreads();

  // ---- phase 3: ctx = P @ V, each wave does its 256 keys, reduce via LDS
  const bf16_t* vbase = Vtb + ((size_t)((b * NH + h) * HD + lm) << 11) + wave * 256;
  v8f cc0 = vzero8(), cc1 = vzero8(), cc2 = vzero8(), cc3 = vzero8();

#pragma unroll
  for (int ks = 0; ks < 8; ++ks) {
    const v16bf vb0 = *(const v16bf*)(vbase + ((size_t)0 * 16 << 11) + ks * 32 + half * 16);
    const v16bf vb1 = *(const v16bf*)(vbase + ((size_t)1 * 16 << 11) + ks * 32 + half * 16);
    const v16bf vb2 = *(const v16bf*)(vbase + ((size_t)2 * 16 << 11) + ks * 32 + half * 16);
    const v16bf vb3 = *(const v16bf*)(vbase + ((size_t)3 * 16 << 11) + ks * 32 + half * 16);
    v16bf a;
    const float* pr = &s_sc[lm * 2052 + wave * 256 + ks * 32];
#pragma unroll
    for (int i = 0; i < 8; ++i) {
      a[i]     = (bf16_t)pr[half * 16 + i];
      a[8 + i] = (bf16_t)pr[half * 16 + 8 + i];
    }
    cc0 = wmma_bf16(a, vb0, cc0);
    cc1 = wmma_bf16(a, vb1, cc1);
    cc2 = wmma_bf16(a, vb2, cc2);
    cc3 = wmma_bf16(a, vb3, cc3);
  }

  auto red_one = [&](const v8f& c, int t) {
#pragma unroll
    for (int rr = 0; rr < 8; ++rr)
      atomicAdd(&s_ctx[(half * 8 + rr) * 64 + t * 16 + lm], c[rr]);
  };
  red_one(cc0, 0); red_one(cc1, 1); red_one(cc2, 2); red_one(cc3, 3);
  __syncthreads();

  for (int i = tid; i < 16 * 64; i += 256) {
    const int m = i >> 6, d = i & 63;
    Ctxb[(size_t)(b * SEQ + q0 + m) * HID + h * HD + d] = (bf16_t)s_ctx[i];
  }
}

// ---------------------------------------------------------------- out proj
// h = ctx @ Wo^T + bo + residual (f32 out). grid(16, 32), block 256.
__global__ __launch_bounds__(256)
void oproj_kernel(const bf16_t* __restrict__ Cb, const bf16_t* __restrict__ Wob,
                  const float* __restrict__ bo, const float* __restrict__ X,
                  float* __restrict__ Hout) {
  const int lane = threadIdx.x & 31;
  const int wave = threadIdx.x >> 5;
  const int half = lane >> 4;
  const int lm   = lane & 15;
  const int n0   = blockIdx.x * 64;
  const int m0   = blockIdx.y * 256 + wave * 32;

  const bf16_t* arow0 = Cb + (size_t)(m0 + lm) * HID;
  const bf16_t* arow1 = arow0 + (size_t)16 * HID;
  const bf16_t* brow0 = Wob + (size_t)(n0 + lm) * HID;

  GEMM_CORE()

  auto store_one = [&](const v8f& c, int t, int ms) {
    const int n = n0 + t * 16 + lm;
    const float bia = bo[n];
#pragma unroll
    for (int r = 0; r < 8; ++r) {
      const int m = m0 + ms * 16 + half * 8 + r;
      Hout[(size_t)m * HID + n] = c[r] + bia + X[(size_t)m * HID + n];
    }
  };
  store_one(c00, 0, 0); store_one(c10, 0, 1);
  store_one(c01, 1, 0); store_one(c11, 1, 1);
  store_one(c02, 2, 0); store_one(c12, 2, 1);
  store_one(c03, 3, 0); store_one(c13, 3, 1);
}

// ---------------------------------------------------------------- layernorm
__global__ __launch_bounds__(256)
void ln_kernel(const float* __restrict__ Hin, const float* __restrict__ g,
               const float* __restrict__ be, float* __restrict__ out) {
  __shared__ float red[256];
  const int row = blockIdx.x;
  const int tid = threadIdx.x;
  const float* hr = Hin + (size_t)row * HID;

  float v[4];
  float s = 0.0f;
#pragma unroll
  for (int i = 0; i < 4; ++i) { v[i] = hr[tid + 256 * i]; s += v[i]; }
  red[tid] = s;
  __syncthreads();
  for (int o = 128; o > 0; o >>= 1) {
    if (tid < o) red[tid] += red[tid + o];
    __syncthreads();
  }
  const float mu = red[0] * (1.0f / HID);
  __syncthreads();

  float s2 = 0.0f;
#pragma unroll
  for (int i = 0; i < 4; ++i) { const float d = v[i] - mu; s2 += d * d; }
  red[tid] = s2;
  __syncthreads();
  for (int o = 128; o > 0; o >>= 1) {
    if (tid < o) red[tid] += red[tid + o];
    __syncthreads();
  }
  const float inv = rsqrtf(red[0] * (1.0f / HID) + 1e-12f);
#pragma unroll
  for (int i = 0; i < 4; ++i) {
    const int c = tid + 256 * i;
    out[(size_t)row * HID + c] = (v[i] - mu) * inv * g[c] + be[c];
  }
}

// ---------------------------------------------------------------- launch
extern "C" void kernel_launch(void* const* d_in, const int* in_sizes, int n_in,
                              void* d_out, int out_size, void* d_ws, size_t ws_size,
                              hipStream_t stream) {
  (void)in_sizes; (void)n_in; (void)out_size; (void)ws_size;
  const float* X     = (const float*)d_in[0];
  const float* amask = (const float*)d_in[1];
  const float* Wq    = (const float*)d_in[2];
  const float* bq    = (const float*)d_in[3];
  const float* Wk    = (const float*)d_in[4];
  const float* bk    = (const float*)d_in[5];
  const float* Wv    = (const float*)d_in[6];
  const float* bv    = (const float*)d_in[7];
  const float* Wo    = (const float*)d_in[8];
  const float* bo    = (const float*)d_in[9];
  const float* ln_g  = (const float*)d_in[10];
  const float* ln_b  = (const float*)d_in[11];

  char* ws = (char*)d_ws;
  bf16_t* Xb   = (bf16_t*)(ws + 0);           // 16 MB
  bf16_t* Wqb  = (bf16_t*)(ws + 16777216);    // 2 MB
  bf16_t* Wkb  = (bf16_t*)(ws + 18874368);    // 2 MB
  bf16_t* Wvb  = (bf16_t*)(ws + 20971520);    // 2 MB
  bf16_t* Wob  = (bf16_t*)(ws + 23068672);    // 2 MB
  bf16_t* Qb   = (bf16_t*)(ws + 25165824);    // 16 MB
  bf16_t* Kb   = (bf16_t*)(ws + 41943040);    // 16 MB
  bf16_t* Vtb  = (bf16_t*)(ws + 58720256);    // 16 MB [B,H,D,S]
  bf16_t* Ctxb = (bf16_t*)(ws + 75497472);    // 16 MB
  float*  Hf   = (float*)(ws + 92274688);     // 32 MB -> 126 MB total

  float* out_f  = (float*)d_out;                         // [8192,1024]
  float* probs  = (float*)d_out + (size_t)NTOK * HID;    // [4,16,2048,2048]

  // 1) fp32 -> bf16 converts (vectorized x4)
  f32_to_bf16_kernel<<<(NTOK * HID / 4) / 256, 256, 0, stream>>>(
      (const v4f*)X, (v4bf*)Xb, NTOK * HID / 4);
  f32_to_bf16_kernel<<<(HID * HID / 4) / 256, 256, 0, stream>>>(
      (const v4f*)Wq, (v4bf*)Wqb, HID * HID / 4);
  f32_to_bf16_kernel<<<(HID * HID / 4) / 256, 256, 0, stream>>>(
      (const v4f*)Wk, (v4bf*)Wkb, HID * HID / 4);
  f32_to_bf16_kernel<<<(HID * HID / 4) / 256, 256, 0, stream>>>(
      (const v4f*)Wv, (v4bf*)Wvb, HID * HID / 4);
  f32_to_bf16_kernel<<<(HID * HID / 4) / 256, 256, 0, stream>>>(
      (const v4f*)Wo, (v4bf*)Wob, HID * HID / 4);

  // 2) QKV projections (V transposed per head)
  qkv_gemm_kernel<<<dim3(HID / 64, NTOK / 256, 3), 256, 0, stream>>>(
      Xb, Wqb, Wkb, Wvb, bq, bk, bv, Qb, Kb, Vtb);

  // 3) fused scores + softmax + probs + P@V
  attn_kernel<<<dim3(SEQ / 16, NH, BATCH), 256, 0, stream>>>(
      Qb, Kb, Vtb, amask, probs, Ctxb);

  // 4) output projection + residual
  oproj_kernel<<<dim3(HID / 64, NTOK / 256), 256, 0, stream>>>(Ctxb, Wob, bo, X, Hf);

  // 5) layernorm
  ln_kernel<<<NTOK, 256, 0, stream>>>(Hf, ln_g, ln_b, out_f);
}
